// TitansMemoryModule_3393024164108
// MI455X (gfx1250) — matrix operand order repaired and checked by reading, not compiled, not run on hardware
//
#include <hip/hip_runtime.h>
#include <hip/hip_bf16.h>
#include <math.h>

// ---------------------------------------------------------------------------
// Titans memory module for MI455X (gfx1250, wave32, WMMA).
// fp32 in/out; internal 16-bit (f16) operands with fp32 WMMA accumulation.
// ---------------------------------------------------------------------------

typedef __attribute__((ext_vector_type(16))) _Float16 v16h;
typedef __attribute__((ext_vector_type(8)))  _Float16 v8h;
typedef __attribute__((ext_vector_type(8)))  float    v8f;

namespace cfg {
constexpr int Bb  = 4;
constexpr int Ll  = 4096;
constexpr int HID = 1024;
constexpr int Hh  = 16;
constexpr int Dd  = 64;
constexpr int DM  = 128;
constexpr int CS  = 64;
constexpr int NC  = Ll / CS;          // 64 chunks
constexpr int BL  = Bb * Ll;          // 16384 rows
}

#define DEV static __device__ __forceinline__

DEV float sigf(float x)   { return 1.0f / (1.0f + __expf(-x)); }
DEV float siluf(float x)  { return x * sigf(x); }
DEV float silubwdf(float x){ float s = sigf(x); return s + x * s * (1.0f - s); }

// ---- WMMA helpers (wave32; A and B both use contiguous k = 16*(lane/16) + i) -------------------

DEV v8f wmma_f16(v16h a, v16h b, v8f c) {
    return __builtin_amdgcn_wmma_f32_16x16x32_f16(
        /*neg_a=*/false, a, /*neg_b=*/false, b,
        /*c_mod=*/(short)0, c, /*reuse_a=*/false, /*reuse_b=*/false);
}

// A fragment: 16x32 f16, row-major source with leading-dim ld (halves).
// lanes 0-15: row=lane,   K = 0..15
// lanes16-31: row=lane-16,K = 16..31
DEV v16h frag_a_f16(const _Float16* src, int ld) {
    int lane = threadIdx.x & 31;
    int row  = lane & 15;
    int k0   = (lane < 16) ? 0 : 16;
    const _Float16* p = src + row * ld + k0;
    v8h lo = *reinterpret_cast<const v8h*>(p);
    v8h hi = *reinterpret_cast<const v8h*>(p + 8);
    v16h a;
#pragma unroll
    for (int i = 0; i < 8; ++i) { a[i] = lo[i]; a[8 + i] = hi[i]; }
    return a;
}

// B fragment from row-major KxN source (B[k][n] = src[k*ld + n]).
// lane n holds column n; lanes 0-15 -> K 0..15, lanes 16-31 -> K 16..31.
DEV v16h frag_b_kxn_f32(const float* src, int ld) {
    int lane = threadIdx.x & 31;
    int col  = lane & 15;
    int k0   = (lane < 16) ? 0 : 16;
    v16h b;
#pragma unroll
    for (int i = 0; i < 16; ++i) b[i] = (_Float16)src[(k0 + i) * ld + col];
    return b;
}

// B fragment from row-major NxK source (i.e. B = Wᵀ, B[k][n] = src[n*ld + k]).
DEV v16h frag_b_nxk_f16(const _Float16* src, int ld) {
    int lane = threadIdx.x & 31;
    int col  = lane & 15;
    int k0   = (lane < 16) ? 0 : 16;
    const _Float16* p = src + col * ld + k0;
    v8h lo = *reinterpret_cast<const v8h*>(p);
    v8h hi = *reinterpret_cast<const v8h*>(p + 8);
    v16h b;
#pragma unroll
    for (int i = 0; i < 8; ++i) { b[i] = lo[i]; b[8 + i] = hi[i]; }
    return b;
}

DEV v16h frag_b_nxk_f32(const float* src, int ld) {
    int lane = threadIdx.x & 31;
    int col  = lane & 15;
    int k0   = (lane < 16) ? 0 : 16;
    v16h b;
#pragma unroll
    for (int i = 0; i < 16; ++i) b[i] = (_Float16)src[col * ld + k0 + i];
    return b;
}

// ---------------------------------------------------------------------------
// Kernel 1: fp32 -> f16 cast
// ---------------------------------------------------------------------------
__global__ void cast_f32_to_f16(const float* __restrict__ src,
                                _Float16* __restrict__ dst, int n) {
    int i = (blockIdx.x * blockDim.x + threadIdx.x) * 4;
    if (i + 3 < n) {
        float4 v = *reinterpret_cast<const float4*>(src + i);
        dst[i]     = (_Float16)v.x;
        dst[i + 1] = (_Float16)v.y;
        dst[i + 2] = (_Float16)v.z;
        dst[i + 3] = (_Float16)v.w;
    } else {
        for (int j = i; j < n; ++j) dst[j] = (_Float16)src[j];
    }
}

// ---------------------------------------------------------------------------
// Kernel 2: QKV GEMM  out[z] = x @ W[z]ᵀ   (M=BL, N=HID, K=HID)
// one wave per block; each wave owns a 16x64 output strip.
// x and W are L2-resident (32MB + 6MB vs 192MB L2). Fragments are
// double-buffered so next-step loads overlap the current WMMAs.
// __launch_bounds__(32,1): one wave per block, full VGPR file -> no spills.
// ---------------------------------------------------------------------------
__global__ void __launch_bounds__(32, 1)
qkv_gemm(const _Float16* __restrict__ xh,
         const _Float16* __restrict__ wh,
         _Float16* __restrict__ outh) {
    constexpr int KS = cfg::HID / 32;          // 32 k-steps
    const int m0 = blockIdx.x * 16;
    const int n0 = blockIdx.y * 64;
    const int z  = blockIdx.z;
    const _Float16* W = wh + (size_t)z * cfg::HID * cfg::HID;
    _Float16* dst = outh + (size_t)z * cfg::BL * cfg::HID;
    const _Float16* arow = xh + (size_t)m0 * cfg::HID;

    v8f acc[4] = {};

    // prologue: fragments for ks = 0
    v16h a_cur = frag_a_f16(arow, cfg::HID);
    v16h b_cur[4];
#pragma unroll
    for (int j = 0; j < 4; ++j)
        b_cur[j] = frag_b_nxk_f16(W + (size_t)(n0 + j * 16) * cfg::HID, cfg::HID);

    for (int ks = 0; ks < KS; ++ks) {
        const int kn = (ks + 1 < KS) ? ks + 1 : ks;   // clamp on last step
        // issue next-step loads before this step's WMMAs
        v16h a_nxt = frag_a_f16(arow + kn * 32, cfg::HID);
        v16h b_nxt[4];
#pragma unroll
        for (int j = 0; j < 4; ++j)
            b_nxt[j] = frag_b_nxk_f16(W + (size_t)(n0 + j * 16) * cfg::HID + kn * 32,
                                      cfg::HID);
        if (ks + 8 < KS)
            __builtin_prefetch(arow + (ks + 8) * 32, 0, 0);
#pragma unroll
        for (int j = 0; j < 4; ++j)
            acc[j] = wmma_f16(a_cur, b_cur[j], acc[j]);
        a_cur = a_nxt;
#pragma unroll
        for (int j = 0; j < 4; ++j) b_cur[j] = b_nxt[j];
    }

    const int lane = threadIdx.x & 31;
    const int nn   = lane & 15;
    const int mb   = (lane < 16) ? 0 : 8;
#pragma unroll
    for (int j = 0; j < 4; ++j)
#pragma unroll
        for (int r = 0; r < 8; ++r)
            dst[(size_t)(m0 + mb + r) * cfg::HID + n0 + j * 16 + nn] =
                (_Float16)acc[j][r];
}

// ---------------------------------------------------------------------------
// Kernel 3: gate projections: sigmoid(x @ Wᵀ + b), H=16 outputs per gate.
// one block per (b,l) row; x row staged in LDS; wave32 shuffle reductions.
// ---------------------------------------------------------------------------
__global__ void gates_kernel(const float* __restrict__ x,
                             const float* __restrict__ wmd, const float* __restrict__ bmd,
                             const float* __restrict__ wsd, const float* __restrict__ bsd,
                             const float* __restrict__ wlr, const float* __restrict__ blr,
                             float* __restrict__ gmd, float* __restrict__ gsd,
                             float* __restrict__ glr) {
    __shared__ float xs[cfg::HID];
    const int row = blockIdx.x;                 // b*L + l
    const float* xr = x + (size_t)row * cfg::HID;
    for (int i = threadIdx.x; i < cfg::HID; i += blockDim.x) xs[i] = xr[i];
    __syncthreads();

    const int wave = threadIdx.x >> 5;
    const int lane = threadIdx.x & 31;
#pragma unroll
    for (int i = 0; i < 6; ++i) {               // 48 outputs / 8 waves
        const int o = wave * 6 + i;
        const int g = o >> 4, h = o & 15;
        const float* w = (g == 0) ? wmd : (g == 1) ? wsd : wlr;
        const float* bb = (g == 0) ? bmd : (g == 1) ? bsd : blr;
        float s = 0.0f;
        for (int k = lane; k < cfg::HID; k += 32) s += xs[k] * w[h * cfg::HID + k];
        for (int off = 16; off > 0; off >>= 1) s += __shfl_down(s, off, 32);
        if (lane == 0) {
            float sg = sigf(s + bb[h]);
            float* dst = (g == 0) ? gmd : (g == 1) ? gsd : glr;
            dst[(size_t)row * cfg::Hh + h] = (g == 0) ? (1.0f - sg) : sg;
        }
    }
}

// ---------------------------------------------------------------------------
// Kernel 4: recurrent chunk scan. One block per (b,h) sequence, 8 waves.
// All state in LDS (~147KB, fits 320KB/WGP). Per chunk:
//   conv+silu(+rmsnorm) -> Z1=k@W1 -> mems=silu(Z1)@W2 -> gZ1=(g_m@W2ᵀ)*silu'
//   -> suffix-product decayed reductions -> rank-1 fast-weight update
//   -> y = silu(q@W1)@W2
// decayed_scan(d,v,init)[last] = (∏d)·init + Σ_t (∏_{s>t} d_s)·v_t
// ---------------------------------------------------------------------------
constexpr size_t SCAN_SMEM =
    (size_t)(64 * 128 + 128 * 64) * 4 +                 // W1f, W2f (fp32)
    (size_t)(3 * 64 * 128 + 4 * 64 * 64) * 2 +          // Z1h,X1h,gZ1h,gmh,qch,kch,vch
    (size_t)(64 * 3 + 65 * 2 + 128 + 64 + 64 + 128) * 4;// gates/suffix/scan state

__global__ void __launch_bounds__(256, 1)
titans_scan(const _Float16* __restrict__ qraw, const _Float16* __restrict__ kraw,
            const _Float16* __restrict__ vraw,
            const float* __restrict__ cqw, const float* __restrict__ cqb,
            const float* __restrict__ ckw, const float* __restrict__ ckb,
            const float* __restrict__ cvw, const float* __restrict__ cvb,
            const float* __restrict__ qnw, const float* __restrict__ knw,
            const float* __restrict__ gmd, const float* __restrict__ gsd,
            const float* __restrict__ glr,
            const float* __restrict__ W1g, const float* __restrict__ W2g,
            float* __restrict__ out) {
    extern __shared__ char smem[];
    float*    W1f  = (float*)smem;                  // 64 x 128
    float*    W2f  = W1f + 64 * 128;                // 128 x 64
    _Float16* Z1h  = (_Float16*)(W2f + 128 * 64);   // 64 x 128
    _Float16* X1h  = Z1h + 64 * 128;                // 64 x 128
    _Float16* gZ1h = X1h + 64 * 128;                // 64 x 128
    _Float16* gmh  = gZ1h + 64 * 128;               // 64 x 64 (unscaled g_m)
    _Float16* qch  = gmh + 64 * 64;                 // 64 x 64
    _Float16* kch  = qch + 64 * 64;
    _Float16* vch  = kch + 64 * 64;
    float* md   = (float*)(vch + 64 * 64);
    float* sd   = md + 64;
    float* lrn  = sd + 64;                          // -lr
    float* sufS = lrn + 64;                         // 65
    float* sufM = sufS + 65;                        // 65
    float* W1s  = sufM + 65;                        // 128
    float* W2s  = W1s + 128;                        // 64
    float* kmo  = W2s + 64;                         // 64
    float* xmo  = kmo + 64;                         // 128

    const int bh = blockIdx.x;
    const int b  = bh / cfg::Hh;
    const int h  = bh % cfg::Hh;
    const int tid  = threadIdx.x;
    const int lane = tid & 31;
    const int wave = tid >> 5;
    const int nn   = lane & 15;
    const int mb   = (lane < 16) ? 0 : 8;

    for (int i = tid; i < 64 * 128; i += 256) {
        W1f[i] = W1g[(size_t)h * 64 * 128 + i];
        W2f[i] = W2g[(size_t)h * 128 * 64 + i];
    }
    if (tid < 128) { W1s[tid] = 0.0f; xmo[tid] = 0.0f; }
    if (tid < 64)  { W2s[tid] = 0.0f; kmo[tid] = 0.0f; }
    __syncthreads();

    for (int c = 0; c < cfg::NC; ++c) {
        const int l0 = c * cfg::CS;

        // --- 1: causal conv (K=4) + SiLU for q,k,v chunk -------------------
        for (int e = tid; e < 3 * 64 * 64; e += 256) {
            const int z = e >> 12, r = e & 4095, t = r >> 6, d = r & 63;
            const int ch = h * 64 + d;
            const _Float16* raw; const float *cw, *cb; _Float16* dst;
            if (z == 0)      { raw = qraw; cw = cqw; cb = cqb; dst = qch; }
            else if (z == 1) { raw = kraw; cw = ckw; cb = ckb; dst = kch; }
            else             { raw = vraw; cw = cvw; cb = cvb; dst = vch; }
            const int l = l0 + t;
            float y = cb[ch];
#pragma unroll
            for (int kk = 0; kk < 4; ++kk) {
                const int li = l - 3 + kk;
                if (li >= 0)
                    y += cw[ch * 4 + kk] *
                         (float)raw[((size_t)b * cfg::Ll + li) * cfg::HID + ch];
            }
            dst[t * 64 + d] = (_Float16)siluf(y);
        }
        if (tid < 64) {
            const size_t gi = ((size_t)b * cfg::Ll + l0 + tid) * cfg::Hh + h;
            md[tid] = gmd[gi]; sd[tid] = gsd[gi]; lrn[tid] = -glr[gi];
        }
        __syncthreads();

        // --- 2: RMSNorm rows of q and k ------------------------------------
        if (tid < 64) {
            float ss = 0.0f;
            for (int d = 0; d < 64; ++d) { float v = (float)kch[tid * 64 + d]; ss += v * v; }
            float sc = rsqrtf(ss * (1.0f / 64.0f) + 1e-6f);
            for (int d = 0; d < 64; ++d)
                kch[tid * 64 + d] = (_Float16)(knw[d] * (float)kch[tid * 64 + d] * sc);
            ss = 0.0f;
            for (int d = 0; d < 64; ++d) { float v = (float)qch[tid * 64 + d]; ss += v * v; }
            sc = rsqrtf(ss * (1.0f / 64.0f) + 1e-6f);
            for (int d = 0; d < 64; ++d)
                qch[tid * 64 + d] = (_Float16)(qnw[d] * (float)qch[tid * 64 + d] * sc);
        }
        __syncthreads();

        // --- 3: Z1 = k @ W1 (64x64x128), keep Z1 and X1=silu(Z1) -----------
        for (int tile = wave; tile < 32; tile += 8) {
            const int tm = tile >> 3, tn = tile & 7;
            v8f acc = {};
#pragma unroll
            for (int ks = 0; ks < 2; ++ks) {
                v16h a  = frag_a_f16(kch + tm * 16 * 64 + ks * 32, 64);
                v16h bm = frag_b_kxn_f32(W1f + ks * 32 * 128 + tn * 16, 128);
                acc = wmma_f16(a, bm, acc);
            }
#pragma unroll
            for (int r = 0; r < 8; ++r) {
                const int row = tm * 16 + mb + r, col = tn * 16 + nn;
                const float z = acc[r];
                Z1h[row * 128 + col] = (_Float16)z;
                X1h[row * 128 + col] = (_Float16)siluf(z);
            }
        }
        __syncthreads();

        // --- 4: g_m = X1 @ W2 - v (64x128x64) ------------------------------
        for (int tile = wave; tile < 16; tile += 8) {
            const int tm = tile >> 2, tn = tile & 3;
            v8f acc = {};
#pragma unroll
            for (int ks = 0; ks < 4; ++ks) {
                v16h a  = frag_a_f16(X1h + tm * 16 * 128 + ks * 32, 128);
                v16h bm = frag_b_kxn_f32(W2f + ks * 32 * 64 + tn * 16, 64);
                acc = wmma_f16(a, bm, acc);
            }
#pragma unroll
            for (int r = 0; r < 8; ++r) {
                const int row = tm * 16 + mb + r, col = tn * 16 + nn;
                gmh[row * 64 + col] = (_Float16)(acc[r] - (float)vch[row * 64 + col]);
            }
        }
        __syncthreads();

        // --- 5: gZ1 = (g_m @ W2ᵀ) * silu'(Z1) * (-lr) (64x64x128) ----------
        for (int tile = wave; tile < 32; tile += 8) {
            const int tm = tile >> 3, tn = tile & 7;
            v8f acc = {};
#pragma unroll
            for (int ks = 0; ks < 2; ++ks) {
                v16h a  = frag_a_f16(gmh + tm * 16 * 64 + ks * 32, 64);
                v16h bm = frag_b_nxk_f32(W2f + (tn * 16) * 64 + ks * 32, 64);
                acc = wmma_f16(a, bm, acc);
            }
#pragma unroll
            for (int r = 0; r < 8; ++r) {
                const int row = tm * 16 + mb + r, col = tn * 16 + nn;
                const float v = acc[r] * silubwdf((float)Z1h[row * 128 + col]) * lrn[row];
                gZ1h[row * 128 + col] = (_Float16)v;
            }
        }
        __syncthreads();

        // --- 6: suffix products of decays ----------------------------------
        if (tid == 0) {
            sufS[64] = 1.0f; sufM[64] = 1.0f;
            for (int s = 63; s >= 0; --s) {
                sufS[s] = sufS[s + 1] * sd[s];
                sufM[s] = sufM[s + 1] * md[s];
            }
        }
        __syncthreads();

        // --- 7: decayed column reductions ----------------------------------
        if (tid < 128) {
            const int j = tid;
            float a1 = sufS[0] * W1s[j];
            float a2 = sufM[0] * xmo[j];
            for (int t = 0; t < 64; ++t) {
                a1 += sufS[t + 1] * (float)gZ1h[t * 128 + j];
                a2 += sufM[t + 1] * (float)X1h[t * 128 + j];
            }
            W1s[j] = a1; xmo[j] = a2;
        } else if (tid < 192) {
            const int d = tid - 128;
            float a1 = sufS[0] * W2s[d];
            for (int t = 0; t < 64; ++t)
                a1 += sufS[t + 1] * (lrn[t] * (float)gmh[t * 64 + d]);
            W2s[d] = a1;
        } else {
            const int d = tid - 192;
            float a1 = sufM[0] * kmo[d];
            for (int t = 0; t < 64; ++t)
                a1 += sufM[t + 1] * (float)kch[t * 64 + d];
            kmo[d] = a1;
        }
        __syncthreads();

        // --- 8: rank-1 fast-weight updates ---------------------------------
        const float mprod = sufM[0];
        for (int i = tid; i < 8192; i += 256) {
            W1f[i] = mprod * W1f[i] + kmo[i >> 7] * W1s[i & 127];
            W2f[i] = mprod * W2f[i] + xmo[i >> 6] * W2s[i & 63];
        }
        __syncthreads();

        // --- 9: Xq = silu(q @ W1new) ---------------------------------------
        for (int tile = wave; tile < 32; tile += 8) {
            const int tm = tile >> 3, tn = tile & 7;
            v8f acc = {};
#pragma unroll
            for (int ks = 0; ks < 2; ++ks) {
                v16h a  = frag_a_f16(qch + tm * 16 * 64 + ks * 32, 64);
                v16h bm = frag_b_kxn_f32(W1f + ks * 32 * 128 + tn * 16, 128);
                acc = wmma_f16(a, bm, acc);
            }
#pragma unroll
            for (int r = 0; r < 8; ++r) {
                const int row = tm * 16 + mb + r, col = tn * 16 + nn;
                X1h[row * 128 + col] = (_Float16)siluf(acc[r]);
            }
        }
        __syncthreads();

        // --- 10: y = Xq @ W2new -> global output ---------------------------
        for (int tile = wave; tile < 16; tile += 8) {
            const int tm = tile >> 2, tn = tile & 3;
            v8f acc = {};
#pragma unroll
            for (int ks = 0; ks < 4; ++ks) {
                v16h a  = frag_a_f16(X1h + tm * 16 * 128 + ks * 32, 128);
                v16h bm = frag_b_kxn_f32(W2f + ks * 32 * 64 + tn * 16, 64);
                acc = wmma_f16(a, bm, acc);
            }
#pragma unroll
            for (int r = 0; r < 8; ++r) {
                const int row = tm * 16 + mb + r, col = tn * 16 + nn;
                out[((size_t)(b * cfg::Ll + l0 + row)) * cfg::HID + h * 64 + col] = acc[r];
            }
        }
        __syncthreads();
    }
}

// ---------------------------------------------------------------------------
extern "C" void kernel_launch(void* const* d_in, const int* in_sizes, int n_in,
                              void* d_out, int out_size, void* d_ws, size_t ws_size,
                              hipStream_t stream) {
    (void)in_sizes; (void)n_in; (void)out_size; (void)ws_size;
    using namespace cfg;

    const float* x    = (const float*)d_in[0];
    const float* Wq   = (const float*)d_in[1];
    const float* Wk   = (const float*)d_in[2];
    const float* Wv   = (const float*)d_in[3];
    const float* cqw  = (const float*)d_in[4];
    const float* cqb  = (const float*)d_in[5];
    const float* ckw  = (const float*)d_in[6];
    const float* ckb  = (const float*)d_in[7];
    const float* cvw  = (const float*)d_in[8];
    const float* cvb  = (const float*)d_in[9];
    const float* qnw  = (const float*)d_in[10];
    const float* knw  = (const float*)d_in[11];
    const float* wmd  = (const float*)d_in[12];
    const float* bmd  = (const float*)d_in[13];
    const float* wsd  = (const float*)d_in[14];
    const float* bsd  = (const float*)d_in[15];
    const float* wlr  = (const float*)d_in[16];
    const float* blr  = (const float*)d_in[17];
    const float* W1g  = (const float*)d_in[18];
    const float* W2g  = (const float*)d_in[19];
    float* out        = (float*)d_out;

    // workspace carve (~137 MiB total)
    size_t off = 0;
    auto carve = [&](size_t bytes) {
        void* p = (char*)d_ws + off;
        off += (bytes + 255) & ~(size_t)255;
        return p;
    };
    _Float16* xh   = (_Float16*)carve((size_t)BL * HID * sizeof(_Float16));
    _Float16* wh   = (_Float16*)carve((size_t)3 * HID * HID * sizeof(_Float16));
    _Float16* qkvh = (_Float16*)carve((size_t)3 * BL * HID * sizeof(_Float16));
    float* gmd     = (float*)carve((size_t)BL * Hh * sizeof(float));
    float* gsd     = (float*)carve((size_t)BL * Hh * sizeof(float));
    float* glr     = (float*)carve((size_t)BL * Hh * sizeof(float));

    // 1) casts to f16
    {
        int n = BL * HID;
        cast_f32_to_f16<<<n / 1024, 256, 0, stream>>>(x, xh, n);
        int nw = HID * HID;
        cast_f32_to_f16<<<nw / 1024, 256, 0, stream>>>(Wq, wh + 0 * (size_t)nw, nw);
        cast_f32_to_f16<<<nw / 1024, 256, 0, stream>>>(Wk, wh + 1 * (size_t)nw, nw);
        cast_f32_to_f16<<<nw / 1024, 256, 0, stream>>>(Wv, wh + 2 * (size_t)nw, nw);
    }

    // 2) q,k,v = x @ Wᵀ via WMMA  (grid: 1024 M-tiles x 16 N-strips x 3 mats)
    qkv_gemm<<<dim3(BL / 16, HID / 64, 3), 32, 0, stream>>>(xh, wh, qkvh);

    // 3) gates
    gates_kernel<<<BL, 256, 0, stream>>>(x, wmd, bmd, wsd, bsd, wlr, blr,
                                         gmd, gsd, glr);

    // 4) recurrent chunk scan: one block per (b,h)
    titans_scan<<<Bb * Hh, 256, SCAN_SMEM, stream>>>(
        qkvh, qkvh + (size_t)BL * HID, qkvh + 2 * (size_t)BL * HID,
        cqw, cqb, ckw, ckb, cvw, cvb, qnw, knw,
        gmd, gsd, glr, W1g, W2g, out);
}
